// Encoder_70798240907550
// MI455X (gfx1250) — compile-verified, matrix operand order checked by reading, not run on hardware
//
#include <hip/hip_runtime.h>
#include <stdint.h>

#define NLc 6
#define Bc  8
#define Tc  512
#define Cn  512
#define Hc  8
#define DKc 64
#define FCc 2048
#define Kc  3
#define Wc  4
#define NDc 9           // 2W+1 rel-pos diagonals
#define PADr 1
#define TPRc (Tc + 2*PADr)   // 514 padded rows (t dimension)
#define USE_TDM 1

typedef __bf16 bf16;
typedef __attribute__((ext_vector_type(16))) __bf16 v16bf;
typedef __attribute__((ext_vector_type(8)))  float  v8f;
typedef unsigned int u32x4 __attribute__((ext_vector_type(4)));
typedef int          i32x4 __attribute__((ext_vector_type(4)));
typedef int          i32x8 __attribute__((ext_vector_type(8)));

union AF { v16bf v; uint4 q[2]; bf16 h[16]; };
union CF { v8f  v; float f[8]; };

__device__ __forceinline__ float bf2f(bf16 u) {
  unsigned short s = __builtin_bit_cast(unsigned short, u);
  unsigned x = ((unsigned)s) << 16;
  return __builtin_bit_cast(float, x);
}
__device__ __forceinline__ bf16 f2bf(float f) {
  unsigned x = __builtin_bit_cast(unsigned, f);
  unsigned r = (x + 0x7FFFu + ((x >> 16) & 1u)) >> 16;
  return __builtin_bit_cast(bf16, (unsigned short)r);
}
__device__ __forceinline__ v8f wmma_bf(const AF& a, const AF& b, v8f c) {
  return __builtin_amdgcn_wmma_f32_16x16x32_bf16(false, a.v, false, b.v,
                                                 (short)0, c, false, false);
}

#if USE_TDM
// TDM: DMA a 2D bf16 tile [rows x rowlen] (row stride `rowstride` elements)
// from global memory into LDS at byte offset lds_off. D# per CDNA5 ISA ch.8.
__device__ __forceinline__ void tdm_load_tile_b16(const bf16* gsrc, unsigned lds_off,
                                                  int rows, int rowlen, int rowstride)
{
  unsigned long long ga = (unsigned long long)(uintptr_t)gsrc;
  u32x4 g0;
  g0[0] = 1u;                                   // count=1, user descriptor
  g0[1] = lds_off;                              // D#.lds_addr
  g0[2] = (unsigned)(ga & 0xFFFFFFFFu);         // D#.global_addr[31:0]
  g0[3] = (unsigned)((ga >> 32) & 0x01FFFFFFu)  // D#.global_addr[56:32]
        | (2u << 30);                           // D#.type = 2 (image)
  i32x8 g1;
  g1[0] = (1 << 16);                            // data_size=1 (2 bytes), no multicast
  g1[1] = (rowlen & 0xFFFF) << 16;              // tensor_dim0[15:0] @ bits63:48
  g1[2] = ((unsigned)rowlen >> 16)              // tensor_dim0[31:16]
        | ((rows & 0xFFFF) << 16);              // tensor_dim1[15:0]
  g1[3] = ((unsigned)rows >> 16)                // tensor_dim1[31:16]
        | ((rowlen & 0xFFFF) << 16);            // tile_dim0
  g1[4] = rows & 0xFFFF;                        // tile_dim1 (tile_dim2 = 0)
  g1[5] = rowstride;                            // tensor_dim0_stride[31:0]
  g1[6] = 0;                                    // stride hi / dim1_stride lo
  g1[7] = 0;
  i32x4 z4 = {0, 0, 0, 0};
#if __clang_major__ >= 23
  i32x8 z8 = {0, 0, 0, 0, 0, 0, 0, 0};
  __builtin_amdgcn_tensor_load_to_lds(g0, g1, z4, z4, z8, 0);
#else
  __builtin_amdgcn_tensor_load_to_lds(g0, g1, z4, z4, 0);
#endif
}
#endif

// ---------------------------------------------------------------------------
// GEMM: D[b, t, oc] (f32, +=opt) = sum_c A[b, t, c](bf16) * W[oc, c](bf16)
// Block 256 thr = 8 waves -> 128(t) x 128(oc) tile; wave = 32x64 (2x4 frags,
// 8 WMMA / 32-deep K step). All fragment loads contiguous 16B per lane.
// ---------------------------------------------------------------------------
__global__ __launch_bounds__(256, 2)
void k_gemm(const bf16* __restrict__ A, long long aBatch, int lda,
            const bf16* __restrict__ Wt, int ldw,
            float* __restrict__ D, long long dBatch, int OC,
            int Kd, int accum)
{
  const int b    = blockIdx.z;
  const bf16* Ab = A + (long long)b * aBatch;
  float*      Db = D + (long long)b * dBatch;
  const int wave = threadIdx.x >> 5, lane = threadIdx.x & 31;
  const int r16  = lane & 15, hi = lane >> 4, mh = hi * 8;
  const int r0   = blockIdx.x * 128 + (wave >> 1) * 32;
  const int oc0  = blockIdx.y * 128 + (wave & 1) * 64;

  CF acc[2][4];
  for (int i = 0; i < 2; ++i) for (int j = 0; j < 4; ++j)
    for (int r = 0; r < 8; ++r) acc[i][j].f[r] = 0.f;

  const bf16* ar0 = Ab + (long long)(r0 + r16) * lda + mh;
  const bf16* ar1 = ar0 + 16LL * lda;
  const bf16* br0 = Wt + (long long)(oc0 + r16) * ldw + mh;
  const bf16* br1 = br0 + 16LL * ldw;
  const bf16* br2 = br0 + 32LL * ldw;
  const bf16* br3 = br0 + 48LL * ldw;

  for (int k0 = 0; k0 < Kd; k0 += 32) {
    AF a0, a1, b0, b1, b2, b3;
    a0.q[0] = *(const uint4*)(ar0 + k0);  a0.q[1] = *(const uint4*)(ar0 + k0 + 16);
    a1.q[0] = *(const uint4*)(ar1 + k0);  a1.q[1] = *(const uint4*)(ar1 + k0 + 16);
    b0.q[0] = *(const uint4*)(br0 + k0);  b0.q[1] = *(const uint4*)(br0 + k0 + 16);
    b1.q[0] = *(const uint4*)(br1 + k0);  b1.q[1] = *(const uint4*)(br1 + k0 + 16);
    b2.q[0] = *(const uint4*)(br2 + k0);  b2.q[1] = *(const uint4*)(br2 + k0 + 16);
    b3.q[0] = *(const uint4*)(br3 + k0);  b3.q[1] = *(const uint4*)(br3 + k0 + 16);
    __builtin_prefetch(ar0 + k0 + 64, 0, 3);   // global_prefetch_b8 (near scope)
    __builtin_prefetch(br0 + k0 + 64, 0, 3);
    __builtin_prefetch(br2 + k0 + 64, 0, 3);
    acc[0][0].v = wmma_bf(a0, b0, acc[0][0].v);
    acc[0][1].v = wmma_bf(a0, b1, acc[0][1].v);
    acc[0][2].v = wmma_bf(a0, b2, acc[0][2].v);
    acc[0][3].v = wmma_bf(a0, b3, acc[0][3].v);
    acc[1][0].v = wmma_bf(a1, b0, acc[1][0].v);
    acc[1][1].v = wmma_bf(a1, b1, acc[1][1].v);
    acc[1][2].v = wmma_bf(a1, b2, acc[1][2].v);
    acc[1][3].v = wmma_bf(a1, b3, acc[1][3].v);
  }
  for (int i = 0; i < 2; ++i)
    for (int j = 0; j < 4; ++j)
      for (int r = 0; r < 8; ++r) {
        int row = r0 + i * 16 + mh + r;
        int col = oc0 + j * 16 + r16;
        float* p = Db + (long long)row * OC + col;
        float v = acc[i][j].f[r];
        if (accum) v += *p;
        *p = v;
      }
}

// ---------------------------------------------------------------------------
// Flash-style attention, one (b,h); wave = 16 query rows; 4 waves share staged
// K/V tiles in LDS (K tile via TDM DMA). Rel-pos window +/-4 on 9 diagonals.
// ---------------------------------------------------------------------------
__global__ __launch_bounds__(128)
void k_attn(const bf16* __restrict__ qb, const bf16* __restrict__ kbuf,
            const bf16* __restrict__ vbuf, const float* __restrict__ xmsk,
            const float* __restrict__ erk_l, const float* __restrict__ erv_l,
            bf16* __restrict__ attnb)
{
  __shared__ __align__(16) bf16 sK[32][64];    // [k_local][d]  (TDM target)
  __shared__ __align__(16) bf16 sVt[64][40];   // [d][k_local], padded pitch
  __shared__ __align__(16) bf16 sP[4][16][32]; // per-wave P tile (C->A relayout)
  __shared__ float sRel[4][16][NDc];
  __shared__ float sDiag[4][16][NDc];
  __shared__ float sM[4][16], sL[4][16];
  __shared__ float sErk[NDc][DKc], sErv[NDc][DKc];

  const int b = blockIdx.z, h = blockIdx.y;
  const int wave = threadIdx.x >> 5, lane = threadIdx.x & 31;
  const int r16 = lane & 15, hi = lane >> 4, mh = hi * 8;
  const int t0 = blockIdx.x * 64 + wave * 16;

  const bf16* qB = qb   + ((long long)b * Tc) * Cn + h * DKc;
  const bf16* kB = kbuf + ((long long)b * Tc) * Cn + h * DKc;
  const bf16* vB = vbuf + ((long long)b * Tc) * Cn + h * DKc;
#if USE_TDM
  const unsigned sK_off = (unsigned)(uintptr_t)&sK[0][0];  // LDS byte offset
#endif

  for (int i = threadIdx.x; i < NDc * DKc; i += 128) {
    sErk[i / DKc][i % DKc] = erk_l[i];
    sErv[i / DKc][i % DKc] = erv_l[i];
  }
  if (lane < 16) { sM[wave][lane] = -3.0e38f; sL[wave][lane] = 0.f; }
  for (int i = lane; i < 16 * NDc; i += 32) sDiag[wave][i / NDc][i % NDc] = -1.0e30f;
  __syncthreads();

  // rel-K logits: rel[row][delta+W] = q_row . erk[delta+W]
  for (int idx = lane; idx < 16 * NDc; idx += 32) {
    int row = idx / NDc, dd = idx % NDc;
    const bf16* qr = qB + (long long)(t0 + row) * Cn;
    float a = 0.f;
    for (int d = 0; d < DKc; ++d) a += bf2f(qr[d]) * sErk[dd][d];
    sRel[wave][row][dd] = a;
  }

  // Q A-fragments (loop invariant), K-chunks d0=0 and d0=32
  AF aq0, aq1;
  {
    const bf16* qr = qB + (long long)(t0 + r16) * Cn + mh;
    aq0.q[0] = *(const uint4*)(qr);      aq0.q[1] = *(const uint4*)(qr + 16);
    aq1.q[0] = *(const uint4*)(qr + 32); aq1.q[1] = *(const uint4*)(qr + 48);
  }
  float mtr[8];
  for (int r = 0; r < 8; ++r) mtr[r] = xmsk[b * Tc + t0 + mh + r];

  CF acc[4];
  for (int dt = 0; dt < 4; ++dt) for (int r = 0; r < 8; ++r) acc[dt].f[r] = 0.f;

  for (int kb0 = 0; kb0 < Tc; kb0 += 32) {
    __syncthreads();
#if USE_TDM
    // K tile [32 x 64] via Tensor Data Mover (wave 0 issues; TENSORcnt tracked)
    if (wave == 0) tdm_load_tile_b16(kB + (long long)kb0 * Cn, sK_off, 32, DKc, Cn);
#endif
    { // cooperative stage: V transposed (and K when TDM disabled)
      int krow = threadIdx.x >> 2, seg = (threadIdx.x & 3) * 16;
#if !USE_TDM
      const bf16* ks = kB + (long long)(kb0 + krow) * Cn + seg;
      *(uint4*)&sK[krow][seg]     = *(const uint4*)ks;
      *(uint4*)&sK[krow][seg + 8] = *(const uint4*)(ks + 8);
#endif
      AF tv;
      const bf16* vs = vB + (long long)(kb0 + krow) * Cn + seg;
      tv.q[0] = *(const uint4*)vs; tv.q[1] = *(const uint4*)(vs + 8);
      for (int j = 0; j < 16; ++j) sVt[seg + j][krow] = tv.h[j];
    }
#if USE_TDM
    if (wave == 0) __builtin_amdgcn_s_wait_tensorcnt(0);
#endif
    __syncthreads();

    // S = Q K^T over 32 keys -> two 16x16 C-frags
    CF sfr[2];
    for (int jj = 0; jj < 2; ++jj) {
      const bf16* kr = &sK[jj * 16 + r16][mh];
      AF b0, b1;
      b0.q[0] = *(const uint4*)(kr);      b0.q[1] = *(const uint4*)(kr + 16);
      b1.q[0] = *(const uint4*)(kr + 32); b1.q[1] = *(const uint4*)(kr + 48);
      for (int r = 0; r < 8; ++r) sfr[jj].f[r] = 0.f;
      sfr[jj].v = wmma_bf(aq0, b0, sfr[jj].v);
      sfr[jj].v = wmma_bf(aq1, b1, sfr[jj].v);
    }
    // rel add + mask + diagonal capture
    for (int jj = 0; jj < 2; ++jj) {
      int ss = kb0 + jj * 16 + r16;
      float ms = xmsk[b * Tc + ss];
      for (int r = 0; r < 8; ++r) {
        int row = mh + r, tt = t0 + row;
        float val = sfr[jj].f[r];
        int delta = ss - tt;
        bool inw = (delta >= -Wc) && (delta <= Wc);
        if (inw) val += sRel[wave][row][delta + Wc];
        if (mtr[r] * ms == 0.f) val = -1.0e4f;
        if (inw) sDiag[wave][row][delta + Wc] = val;
        sfr[jj].f[r] = val;
      }
    }
    // online softmax (16-lane reductions match C-frag row ownership)
    float rscale[8];
    for (int r = 0; r < 8; ++r) {
      int row = mh + r;
      float v = fmaxf(sfr[0].f[r], sfr[1].f[r]);
      for (int o = 1; o < 16; o <<= 1) v = fmaxf(v, __shfl_xor(v, o, 32));
      float mold = sM[wave][row];
      float mnew = fmaxf(mold, v);
      float sc = __expf(mold - mnew);
      float p0 = __expf(sfr[0].f[r] - mnew);
      float p1 = __expf(sfr[1].f[r] - mnew);
      sfr[0].f[r] = p0; sfr[1].f[r] = p1;
      float su = p0 + p1;
      for (int o = 1; o < 16; o <<= 1) su += __shfl_xor(su, o, 32);
      sM[wave][row] = mnew;
      sL[wave][row] = sL[wave][row] * sc + su;
      rscale[r] = sc;
      sP[wave][row][r16]      = f2bf(p0);   // C-layout -> LDS (A relayout)
      sP[wave][row][16 + r16] = f2bf(p1);
    }
    for (int dt = 0; dt < 4; ++dt)
      for (int r = 0; r < 8; ++r) acc[dt].f[r] *= rscale[r];

    // O += P V  (A from sP, B from transposed V)
    AF ap;
    { const bf16* pr = &sP[wave][r16][mh];
      ap.q[0] = *(const uint4*)pr; ap.q[1] = *(const uint4*)(pr + 16); }
    for (int dt = 0; dt < 4; ++dt) {
      const bf16* vr = &sVt[dt * 16 + r16][mh];
      AF bv;
      bv.q[0] = *(const uint4*)vr; bv.q[1] = *(const uint4*)(vr + 16);
      acc[dt].v = wmma_bf(ap, bv, acc[dt].v);
    }
  }

  __syncthreads();
  // finalize diagonal probabilities: p = exp(S - m_final)/l
  for (int idx = lane; idx < 16 * NDc; idx += 32) {
    int row = idx / NDc, dd = idx % NDc;
    sDiag[wave][row][dd] =
        __expf(sDiag[wave][row][dd] - sM[wave][row]) / sL[wave][row];
  }
  __syncthreads();
  // epilogue: normalize + rel-V contribution, store bf16 [B,T,C]
  for (int dt = 0; dt < 4; ++dt) {
    int d = dt * 16 + r16;
    for (int r = 0; r < 8; ++r) {
      int row = mh + r, tt = t0 + row;
      float o = acc[dt].f[r] / sL[wave][row];
      for (int dd = 0; dd < NDc; ++dd) o += sDiag[wave][row][dd] * sErv[dd][d];
      attnb[((long long)b * Tc + tt) * Cn + h * DKc + d] = f2bf(o);
    }
  }
}

// ------------------------- elementwise / utility ---------------------------
__global__ void k_cvt(const float* __restrict__ s, bf16* __restrict__ d, long long n) {
  for (long long i = blockIdx.x * 256LL + threadIdx.x; i < n;
       i += (long long)gridDim.x * 256) d[i] = f2bf(s[i]);
}
// [NL,OC,IC,K] -> [NL][K][OC][IC]
__global__ void k_cvt_conv(const float* __restrict__ s, bf16* __restrict__ d,
                           int OC, int IC, long long n) {
  for (long long i = blockIdx.x * 256LL + threadIdx.x; i < n;
       i += (long long)gridDim.x * 256) {
    int k = (int)(i % Kc);
    long long t = i / Kc;
    int c = (int)(t % IC); t /= IC;
    int o = (int)(t % OC);
    long long l = t / OC;
    d[(((l * Kc + k) * (long long)OC + o) * IC) + c] = f2bf(s[i]);
  }
}
__global__ void k_zero16(bf16* __restrict__ p, long long n) {
  bf16 z = __builtin_bit_cast(bf16, (unsigned short)0);
  for (long long i = blockIdx.x * 256LL + threadIdx.x; i < n;
       i += (long long)gridDim.x * 256) p[i] = z;
}
// x [B,C,T] f32 -> xcur [B,T,C] f32 (masked) + padded bf16
__global__ void k_prep(const float* __restrict__ x, const float* __restrict__ xm,
                       float* __restrict__ xcur, bf16* __restrict__ xpad) {
  long long n = (long long)Bc * Cn * Tc;
  for (long long i = blockIdx.x * 256LL + threadIdx.x; i < n;
       i += (long long)gridDim.x * 256) {
    int t = (int)(i % Tc);
    long long bc = i / Tc;
    int c = (int)(bc % Cn), b = (int)(bc / Cn);
    float v = x[i] * xm[b * Tc + t];
    xcur[((long long)b * Tc + t) * Cn + c] = v;
    xpad[((long long)b * TPRc + PADr + t) * Cn + c] = f2bf(v);
  }
}
__global__ void k_qkvpost(const float* __restrict__ src, const float* __restrict__ bias,
                          float scale, bf16* __restrict__ dst) {
  long long n = (long long)Bc * Tc * Cn;
  for (long long i = blockIdx.x * 256LL + threadIdx.x; i < n;
       i += (long long)gridDim.x * 256) {
    int c = (int)(i % Cn);
    dst[i] = f2bf((src[i] + bias[c]) * scale);
  }
}
__global__ __launch_bounds__(256)
void k_res_ln(const float* xin, const float* __restrict__ yin,
              const float* __restrict__ badd, const float* __restrict__ g,
              const float* __restrict__ be, const float* __restrict__ xm,
              float* xout, bf16* __restrict__ xpad) {
  __shared__ float red[256];
  int bt = blockIdx.x, tid = threadIdx.x;
  int b = bt / Tc, t = bt % Tc;
  long long base = (long long)bt * Cn;
  int c0 = tid, c1 = tid + 256;
  float v0 = xin[base + c0] + yin[base + c0] + badd[c0];
  float v1 = xin[base + c1] + yin[base + c1] + badd[c1];
  red[tid] = v0 + v1; __syncthreads();
  for (int o = 128; o > 0; o >>= 1) { if (tid < o) red[tid] += red[tid + o]; __syncthreads(); }
  float mean = red[0] * (1.0f / Cn);
  __syncthreads();
  float d0 = v0 - mean, d1 = v1 - mean;
  red[tid] = d0 * d0 + d1 * d1; __syncthreads();
  for (int o = 128; o > 0; o >>= 1) { if (tid < o) red[tid] += red[tid + o]; __syncthreads(); }
  float rstd = rsqrtf(red[0] * (1.0f / Cn) + 1e-5f);
  float m = xm[b * Tc + t];
  float n0 = d0 * rstd * g[c0] + be[c0];
  float n1 = d1 * rstd * g[c1] + be[c1];
  xout[base + c0] = n0; xout[base + c1] = n1;
  long long pb = ((long long)b * TPRc + PADr + t) * Cn;
  xpad[pb + c0] = f2bf(n0 * m);
  xpad[pb + c1] = f2bf(n1 * m);
}
__global__ void k_h1post(const float* __restrict__ hf, const float* __restrict__ bf1,
                         const float* __restrict__ xm, bf16* __restrict__ h1b) {
  long long n = (long long)Bc * Tc * FCc;
  for (long long i = blockIdx.x * 256LL + threadIdx.x; i < n;
       i += (long long)gridDim.x * 256) {
    int f = (int)(i % FCc);
    long long bt = i / FCc;
    int t = (int)(bt % Tc), b = (int)(bt / Tc);
    float v = hf[i] + bf1[f];
    v = v > 0.f ? v : 0.f;
    h1b[((long long)b * TPRc + PADr + t) * FCc + f] = f2bf(v * xm[b * Tc + t]);
  }
}
__global__ void k_final(const float* __restrict__ xcur, const float* __restrict__ xm,
                        float* __restrict__ out) {
  long long n = (long long)Bc * Cn * Tc;
  for (long long i = blockIdx.x * 256LL + threadIdx.x; i < n;
       i += (long long)gridDim.x * 256) {
    int t = (int)(i % Tc);
    long long bc = i / Tc;
    int c = (int)(bc % Cn), b = (int)(bc / Cn);
    out[i] = xcur[((long long)b * Tc + t) * Cn + c] * xm[b * Tc + t];
  }
}

static inline unsigned grd(long long n) {
  long long g = (n + 255) / 256;
  if (g < 1) g = 1;
  if (g > 1048576) g = 1048576;
  return (unsigned)g;
}

extern "C" void kernel_launch(void* const* d_in, const int* in_sizes, int n_in,
                              void* d_out, int out_size, void* d_ws, size_t ws_size,
                              hipStream_t stream) {
  (void)in_sizes; (void)n_in; (void)out_size; (void)ws_size;
  const float* x   = (const float*)d_in[0];
  const float* xm  = (const float*)d_in[1];
  const float* Wq  = (const float*)d_in[2];
  const float* bq  = (const float*)d_in[3];
  const float* Wk  = (const float*)d_in[4];
  const float* bk_ = (const float*)d_in[5];
  const float* Wv  = (const float*)d_in[6];
  const float* bv_ = (const float*)d_in[7];
  const float* Wo  = (const float*)d_in[8];
  const float* bo_ = (const float*)d_in[9];
  const float* erk = (const float*)d_in[10];
  const float* erv = (const float*)d_in[11];
  const float* g1  = (const float*)d_in[12];
  const float* be1 = (const float*)d_in[13];
  const float* Wf1 = (const float*)d_in[14];
  const float* bf1 = (const float*)d_in[15];
  const float* Wf2 = (const float*)d_in[16];
  const float* bf2 = (const float*)d_in[17];
  const float* g2  = (const float*)d_in[18];
  const float* be2 = (const float*)d_in[19];
  float* out = (float*)d_out;

  char* base = (char*)d_ws; size_t off = 0;
  auto alloc = [&](size_t bytes) -> char* {
    char* p = base + off; off = (off + bytes + 255) & ~(size_t)255; return p;
  };
  const long long nlin  = (long long)NLc * Cn * Cn;
  const long long nbtc  = (long long)Bc * Tc * Cn;
  const long long nbtf  = (long long)Bc * Tc * FCc;
  bf16*  wqb  = (bf16*)alloc(2 * nlin);
  bf16*  wkb  = (bf16*)alloc(2 * nlin);
  bf16*  wvb  = (bf16*)alloc(2 * nlin);
  bf16*  wob  = (bf16*)alloc(2 * nlin);
  bf16*  wf1t = (bf16*)alloc(2ULL * NLc * Kc * FCc * Cn);
  bf16*  wf2t = (bf16*)alloc(2ULL * NLc * Kc * Cn * FCc);
  float* xcur = (float*)alloc(4 * nbtc);
  bf16*  xpad = (bf16*)alloc(2ULL * Bc * TPRc * Cn);
  float* qf   = (float*)alloc(4 * nbtc);
  float* kf   = (float*)alloc(4 * nbtc);
  float* vf   = (float*)alloc(4 * nbtc);
  bf16*  qbv  = (bf16*)alloc(2 * nbtc);
  bf16*  kbv  = (bf16*)alloc(2 * nbtc);
  bf16*  vbv  = (bf16*)alloc(2 * nbtc);
  bf16*  attnb= (bf16*)alloc(2 * nbtc);
  float* yf   = (float*)alloc(4 * nbtc);
  float* hf   = (float*)alloc(4 * nbtf);
  bf16*  h1b  = (bf16*)alloc(2ULL * Bc * TPRc * FCc);

  // weight conversion (f32 -> bf16, conv weights split by tap k)
  k_cvt<<<grd(nlin), 256, 0, stream>>>(Wq, wqb, nlin);
  k_cvt<<<grd(nlin), 256, 0, stream>>>(Wk, wkb, nlin);
  k_cvt<<<grd(nlin), 256, 0, stream>>>(Wv, wvb, nlin);
  k_cvt<<<grd(nlin), 256, 0, stream>>>(Wo, wob, nlin);
  long long n1 = (long long)NLc * FCc * Cn * Kc;
  long long n2 = (long long)NLc * Cn * FCc * Kc;
  k_cvt_conv<<<grd(n1), 256, 0, stream>>>(Wf1, wf1t, FCc, Cn, n1);
  k_cvt_conv<<<grd(n2), 256, 0, stream>>>(Wf2, wf2t, Cn, FCc, n2);
  k_zero16<<<grd((long long)Bc * TPRc * Cn), 256, 0, stream>>>(xpad, (long long)Bc * TPRc * Cn);
  k_zero16<<<grd((long long)Bc * TPRc * FCc), 256, 0, stream>>>(h1b, (long long)Bc * TPRc * FCc);
  k_prep<<<grd(nbtc), 256, 0, stream>>>(x, xm, xcur, xpad);

  const dim3 gproj(Tc / 128, Cn / 128, Bc);   // 128x128 block tiles
  const dim3 gc1(Tc / 128, FCc / 128, Bc);
  const dim3 gattn(Tc / 64, Hc, Bc);
  const long long aB  = (long long)TPRc * Cn;   // padded activation batch stride
  const long long aBh = (long long)TPRc * FCc;
  const float qscale = 0.125f;                  // 1/sqrt(DK)

  for (int i = 0; i < NLc; ++i) {
    const bf16* wq_l = wqb + (long long)i * Cn * Cn;
    const bf16* wk_l = wkb + (long long)i * Cn * Cn;
    const bf16* wv_l = wvb + (long long)i * Cn * Cn;
    const bf16* wo_l = wob + (long long)i * Cn * Cn;

    // Q/K/V projections: A = padded x rows, B = weight rows
    k_gemm<<<gproj, 256, 0, stream>>>(xpad + (long long)PADr * Cn, aB, Cn,
                                      wq_l, Cn, qf, (long long)Tc * Cn, Cn, Cn, 0);
    k_gemm<<<gproj, 256, 0, stream>>>(xpad + (long long)PADr * Cn, aB, Cn,
                                      wk_l, Cn, kf, (long long)Tc * Cn, Cn, Cn, 0);
    k_gemm<<<gproj, 256, 0, stream>>>(xpad + (long long)PADr * Cn, aB, Cn,
                                      wv_l, Cn, vf, (long long)Tc * Cn, Cn, Cn, 0);
    k_qkvpost<<<grd(nbtc), 256, 0, stream>>>(qf, bq  + (long long)i * Cn, qscale, qbv);
    k_qkvpost<<<grd(nbtc), 256, 0, stream>>>(kf, bk_ + (long long)i * Cn, 1.0f,   kbv);
    k_qkvpost<<<grd(nbtc), 256, 0, stream>>>(vf, bv_ + (long long)i * Cn, 1.0f,   vbv);

    k_attn<<<gattn, 128, 0, stream>>>(qbv, kbv, vbv, xm,
                                      erk + (long long)i * NDc * DKc,
                                      erv + (long long)i * NDc * DKc, attnb);

    k_gemm<<<gproj, 256, 0, stream>>>(attnb, (long long)Tc * Cn, Cn,
                                      wo_l, Cn, yf, (long long)Tc * Cn, Cn, Cn, 0);
    k_res_ln<<<Bc * Tc, 256, 0, stream>>>(xcur, yf, bo_ + (long long)i * Cn,
                                          g1 + (long long)i * Cn, be1 + (long long)i * Cn,
                                          xm, xcur, xpad);
    // FFN conv1 (K=3 taps as accumulated GEMMs against padded rows)
    for (int k = 0; k < Kc; ++k)
      k_gemm<<<gc1, 256, 0, stream>>>(xpad + (long long)(PADr - 1 + k) * Cn, aB, Cn,
                                      wf1t + ((long long)i * Kc + k) * FCc * Cn, Cn,
                                      hf, (long long)Tc * FCc, FCc, Cn, k > 0);
    k_h1post<<<grd(nbtf), 256, 0, stream>>>(hf, bf1 + (long long)i * FCc, xm, h1b);
    for (int k = 0; k < Kc; ++k)
      k_gemm<<<gproj, 256, 0, stream>>>(h1b + (long long)(PADr - 1 + k) * FCc, aBh, FCc,
                                        wf2t + ((long long)i * Kc + k) * Cn * FCc, FCc,
                                        yf, (long long)Tc * Cn, Cn, FCc, k > 0);
    k_res_ln<<<Bc * Tc, 256, 0, stream>>>(xcur, yf, bf2 + (long long)i * Cn,
                                          g2 + (long long)i * Cn, be2 + (long long)i * Cn,
                                          xm, xcur, xpad);
  }
  k_final<<<grd(nbtc), 256, 0, stream>>>(xcur, xm, out);
}
